// LocalAttention_48945447305896
// MI455X (gfx1250) — matrix operand order repaired and checked by reading, not compiled, not run on hardware
//
#include <hip/hip_runtime.h>

typedef __attribute__((ext_vector_type(16))) _Float16 v16h;
typedef __attribute__((ext_vector_type(8)))  _Float16 v8h;
typedef __attribute__((ext_vector_type(8)))  float    v8f;

#define P      7
#define H      6
#define DH     30
#define DIM    180
#define INNER  180
#define BATCH  16
#define LTOT   14336
#define NWIN   8            // windows per workgroup
#define NPOS   (NWIN * P)   // 56 positions, padded to 64 rows
#define NROWS  64
#define KDIM   192          // padded K stride == 6 heads * 32

// f16 weight workspace layout (in halves)
#define WQ_OFF  0
#define WKV_OFF (192 * 192)
#define WO_OFF  (WKV_OFF + 368 * 192)

// ---------------------------------------------------------------------------
// Weight prep: fp32 -> fp16, pad K to 192; Wo gets its K (e) remapped into the
// head-padded layout (h*32 + d, zeros at d=30,31) so the final GEMM can run a
// clean K=192 with zero contributions from the pads.
// ---------------------------------------------------------------------------
__global__ void prep_weights(const float* __restrict__ Wq,
                             const float* __restrict__ Wkv,
                             const float* __restrict__ Wo,
                             _Float16* __restrict__ ws) {
  int idx = blockIdx.x * 256 + threadIdx.x;
  if (idx >= 368 * 192) return;
  int r = idx / 192, c = idx % 192;
  if (r < 192) {
    float vq = (r < INNER && c < DIM) ? Wq[r * DIM + c] : 0.f;
    ws[WQ_OFF + r * 192 + c] = (_Float16)vq;
    int h = c >> 5, d = c & 31;
    float vo = (r < DIM && d < DH) ? Wo[r * INNER + h * DH + d] : 0.f;
    ws[WO_OFF + r * 192 + c] = (_Float16)vo;
  }
  float vkv = (r < 2 * INNER && c < DIM) ? Wkv[r * DIM + c] : 0.f;
  ws[WKV_OFF + r * 192 + c] = (_Float16)vkv;
}

// ---------------------------------------------------------------------------
// WMMA fragment helpers (CDNA5 wave32 VGPR layouts, 16x16x32 f16)
// A (16x32): lane = m&15; k = kbase + (h&7) + 16*(h>>3) + 8*(lane>>4)
// B (32x16): lane = n&15; k = kbase + (h&7) + 16*(h>>3) + 8*(lane>>4)
// C (16x16): m = r + 8*(lane>>4), n = lane&15
// ---------------------------------------------------------------------------
__device__ __forceinline__ v16h load_a16(const _Float16* base, int stride,
                                         int mrow, int kbase, int lane) {
  int grp = lane >> 4;
  const _Float16* p = base + (size_t)mrow * stride + kbase + 8 * grp;
  v8h lo = *(const v8h*)p;
  v8h hi = *(const v8h*)(p + 16);
  v16h a;
#pragma unroll
  for (int i = 0; i < 8; ++i) { a[i] = lo[i]; a[i + 8] = hi[i]; }
  return a;
}

__device__ __forceinline__ v16h load_b16(const _Float16* base, int stride,
                                         int nrow, int kbase, int lane) {
  int grp = lane >> 4;
  const _Float16* p = base + (size_t)nrow * stride + kbase + 8 * grp;
  v8h lo = *(const v8h*)p;
  v8h hi = *(const v8h*)(p + 16);
  v16h b;
#pragma unroll
  for (int i = 0; i < 8; ++i) { b[i] = lo[i]; b[i + 8] = hi[i]; }
  return b;
}

__device__ __forceinline__ v8f wmma_f16(v16h a, v16h b, v8f c) {
  return __builtin_amdgcn_wmma_f32_16x16x32_f16(false, a, false, b,
                                                (short)0, c, false, false);
}

// ---------------------------------------------------------------------------
// Fused local attention: x tile -> Q/KV GEMM -> windowed attention -> O GEMM
// One workgroup = one batch x 8 windows (56 positions).
// ---------------------------------------------------------------------------
__global__ __launch_bounds__(256) void local_attn_fused(
    const float* __restrict__ fmap, const _Float16* __restrict__ ws,
    const float* __restrict__ bo, float* __restrict__ out) {
  extern __shared__ _Float16 lds[];
  _Float16* xbuf = lds;                      // [64][192] x tile, reused as attn-out
  _Float16* qbuf = lds + 1 * NROWS * KDIM;   // [64][192] q, head-padded (h*32+d)
  _Float16* kbuf = lds + 2 * NROWS * KDIM;   // [64][192] k
  _Float16* vbuf = lds + 3 * NROWS * KDIM;   // [64][192] v
  _Float16* scr  = lds + 4 * NROWS * KDIM;   // 8 waves * 16x16 attn scratch

  const int tid  = threadIdx.x;
  const int lane = tid & 31;
  const int wave = tid >> 5;
  const int grp  = lane >> 4;
  const int g    = blockIdx.x;               // window group
  const int bb   = blockIdx.y;               // batch
  const long posBase = (long)g * NPOS;

  const _Float16* WqH  = ws + WQ_OFF;
  const _Float16* WkvH = ws + WKV_OFF;
  const _Float16* WoH  = ws + WO_OFF;

  // Zero all LDS (pads must be finite zeros: NaN*0 = NaN in WMMA).
  {
    v8h z = {};
    const int total = 4 * NROWS * KDIM + 8 * 256;
    for (int i = tid * 8; i < total; i += 256 * 8) *(v8h*)(lds + i) = z;
  }
  __syncthreads();

  // Load x tile: fmap[b][c][pos] -> xbuf[n][c] (f16), coalesced along n.
  for (int idx = tid; idx < DIM * NPOS; idx += 256) {
    int c = idx / NPOS, n = idx % NPOS;
    float v = fmap[((size_t)bb * DIM + c) * LTOT + posBase + n];
    xbuf[n * KDIM + c] = (_Float16)v;
  }
  __syncthreads();

  // ---- Q projection: M=192 (12 tiles) x N=64 (4 tiles), K=192 --------------
  for (int tile = wave; tile < 12 * 4; tile += 8) {
    int mt = tile >> 2, nt = tile & 3;
    int mrow = mt * 16 + (lane & 15);
    int nrow = nt * 16 + (lane & 15);
    __builtin_prefetch(WqH + (size_t)mrow * 192);
    v8f acc = {};
#pragma unroll
    for (int kk = 0; kk < 6; ++kk) {
      v16h a = load_a16(WqH, 192, mrow, kk * 32, lane);
      v16h b = load_b16(xbuf, KDIM, nrow, kk * 32, lane);
      acc = wmma_f16(a, b, acc);
    }
    int n = nt * 16 + (lane & 15);
#pragma unroll
    for (int r = 0; r < 8; ++r) {
      int m = mt * 16 + r + 8 * grp;
      if (m < INNER) qbuf[n * KDIM + (m / DH) * 32 + (m % DH)] = (_Float16)acc[r];
    }
  }

  // ---- KV projection: M=368 (23 tiles) x N=64 (4 tiles), K=192 ------------
  for (int tile = wave; tile < 23 * 4; tile += 8) {
    int mt = tile >> 2, nt = tile & 3;
    int mrow = mt * 16 + (lane & 15);
    int nrow = nt * 16 + (lane & 15);
    __builtin_prefetch(WkvH + (size_t)mrow * 192);
    v8f acc = {};
#pragma unroll
    for (int kk = 0; kk < 6; ++kk) {
      v16h a = load_a16(WkvH, 192, mrow, kk * 32, lane);
      v16h b = load_b16(xbuf, KDIM, nrow, kk * 32, lane);
      acc = wmma_f16(a, b, acc);
    }
    int n = nt * 16 + (lane & 15);
#pragma unroll
    for (int r = 0; r < 8; ++r) {
      int m = mt * 16 + r + 8 * grp;
      if (m < INNER) {
        kbuf[n * KDIM + (m / DH) * 32 + (m % DH)] = (_Float16)acc[r];
      } else if (m < 2 * INNER) {
        int e = m - INNER;
        vbuf[n * KDIM + (e / DH) * 32 + (e % DH)] = (_Float16)acc[r];
      }
    }
  }
  __syncthreads();

  // ---- Windowed attention: one (window, head) task per wave iteration -----
  const float scale = 0.18257419f;  // 1/sqrt(30)
  _Float16* wscr = scr + wave * 256;
  for (int task = wave; task < NWIN * H; task += 8) {
    int w = task / H, head = task % H;
    int n0 = w * P;
    // scores = q(7x30 pad 16x32) . k^T(30x7 pad 32x16): one WMMA
    v16h qa = load_a16(qbuf, KDIM, n0 + (lane & 15), head * 32, lane);
    v16h kb = load_b16(kbuf, KDIM, n0 + (lane & 15), head * 32, lane);
    v8f sc = {};
    sc = wmma_f16(qa, kb, sc);

    // softmax over the 7 valid columns (lanes 0..6, group 0), rows 0..6
    float pr[8];
    int jcol = lane & 15;
#pragma unroll
    for (int r = 0; r < 8; ++r) {
      bool valid = (grp == 0) && (jcol < P) && (r < P);
      float v = sc[r] * scale;
      float mx = valid ? v : -1e30f;
      mx = fmaxf(mx, __shfl_xor(mx, 1, 32));
      mx = fmaxf(mx, __shfl_xor(mx, 2, 32));
      mx = fmaxf(mx, __shfl_xor(mx, 4, 32));
      float ex = valid ? __expf(v - mx) : 0.f;
      float sm = ex;
      sm += __shfl_xor(sm, 1, 32);
      sm += __shfl_xor(sm, 2, 32);
      sm += __shfl_xor(sm, 4, 32);
      pr[r] = valid ? (ex / sm) : 0.f;
    }
    // park attn in wave-private LDS scratch (16x16, zero outside 7x7)
#pragma unroll
    for (int r = 0; r < 8; ++r) {
      int i = r + 8 * grp;
      wscr[i * 16 + jcol] = (_Float16)pr[r];
    }
    // attn as A fragment (K=7 padded to 32 with zeros)
    v16h aa;
    {
      const _Float16* pA = wscr + (lane & 15) * 16 + 8 * grp;
      v8h lo = *(const v8h*)pA;
#pragma unroll
      for (int i = 0; i < 8; ++i) { aa[i] = lo[i]; aa[i + 8] = (_Float16)0.f; }
    }
    // out = attn @ v : two N tiles of 16 (d = 0..15, 16..31 with 30,31 pad)
#pragma unroll
    for (int t = 0; t < 2; ++t) {
      int dbase = t * 16;
      int dcol = lane & 15;
      v16h vb;
#pragma unroll
      for (int h2 = 0; h2 < 16; ++h2) {
        int jj = 8 * grp + (h2 & 7) + 16 * (h2 >> 3);
        vb[h2] = (jj < P) ? vbuf[(n0 + jj) * KDIM + head * 32 + dbase + dcol]
                          : (_Float16)0.f;
      }
      v8f oc = {};
      oc = wmma_f16(aa, vb, oc);
#pragma unroll
      for (int r = 0; r < 8; ++r) {
        int i = r + 8 * grp;
        if (i < P) {
          int dd = dbase + dcol;
          xbuf[(n0 + i) * KDIM + head * 32 + dd] =
              (dd < DH) ? (_Float16)oc[r] : (_Float16)0.f;
        }
      }
    }
  }
  __syncthreads();

  // ---- Output projection + bias: Wo16(192x192) @ out(192x64) -------------
  for (int tile = wave; tile < 12 * 4; tile += 8) {
    int mt = tile >> 2, nt = tile & 3;
    int mrow = mt * 16 + (lane & 15);
    int nrow = nt * 16 + (lane & 15);
    __builtin_prefetch(WoH + (size_t)mrow * 192);
    v8f acc = {};
#pragma unroll
    for (int kk = 0; kk < 6; ++kk) {
      v16h a = load_a16(WoH, 192, mrow, kk * 32, lane);
      v16h b = load_b16(xbuf, KDIM, nrow, kk * 32, lane);
      acc = wmma_f16(a, b, acc);
    }
    int n = nt * 16 + (lane & 15);
    if (n < NPOS) {
#pragma unroll
      for (int r = 0; r < 8; ++r) {
        int m = mt * 16 + r + 8 * grp;
        if (m < DIM)
          out[((size_t)bb * DIM + m) * LTOT + posBase + n] = acc[r] + bo[m];
      }
    }
  }
}

// ---------------------------------------------------------------------------
extern "C" void kernel_launch(void* const* d_in, const int* in_sizes, int n_in,
                              void* d_out, int out_size, void* d_ws,
                              size_t ws_size, hipStream_t stream) {
  (void)in_sizes; (void)n_in; (void)out_size; (void)ws_size;
  const float* fmap = (const float*)d_in[0];
  const float* Wq   = (const float*)d_in[1];
  const float* Wkv  = (const float*)d_in[2];
  const float* Wo   = (const float*)d_in[3];
  const float* bo   = (const float*)d_in[4];
  float* out = (float*)d_out;
  _Float16* ws = (_Float16*)d_ws;   // ~289 KB of f16 weights

  {
    int total = 368 * 192;
    prep_weights<<<(total + 255) / 256, 256, 0, stream>>>(Wq, Wkv, Wo, ws);
  }
  dim3 grid(LTOT / P / NWIN, BATCH);  // (256, 16) workgroups
  size_t ldsBytes = (size_t)(4 * NROWS * KDIM + 8 * 256) * sizeof(_Float16);
  local_attn_fused<<<grid, 256, ldsBytes, stream>>>(fmap, ws, bo, out);
}
